// Triton_17377437680119
// MI455X (gfx1250) — matrix operand order repaired and checked by reading, not compiled, not run on hardware
//
// MI455X (gfx1250) mask-gated dual SwiGLU FFN.
// Strategy: stable mask compaction -> per-branch bf16 WMMA GEMMs with fp32
// accumulate; weights pre-packed into WMMA B-fragment tile layout; LDS staging
// via global_load_async_to_lds_b128 with double buffering + s_wait_asynccnt.
#include <hip/hip_runtime.h>

constexpr int kH   = 2048;   // hidden
constexpr int kI   = 8192;   // intermediate
constexpr int kTok = 8192;   // B*S

typedef __attribute__((ext_vector_type(8)))  unsigned short us8;
typedef __attribute__((ext_vector_type(16))) unsigned short us16;
typedef __attribute__((ext_vector_type(16))) __bf16         v16bf;
typedef __attribute__((ext_vector_type(8)))  float          v8f;

__device__ __forceinline__ unsigned short f2bf(float f) {
  unsigned u = __builtin_bit_cast(unsigned, f);
  u += 0x7FFFu + ((u >> 16) & 1u);          // round-to-nearest-even
  return (unsigned short)(u >> 16);
}

__device__ __forceinline__ unsigned lds_off(const void* p) {
  // low 32 bits of the generic address of an LDS object == LDS byte offset
  return (unsigned)(size_t)p;
}

__device__ __forceinline__ void async_ld_b128(unsigned lds_addr, unsigned voff,
                                              unsigned long long sbase) {
  // GVS mode: saddr = 64-bit base, vaddr = 32-bit per-lane byte offset,
  // vdst = LDS byte address. Tracked by ASYNCcnt.
  asm volatile("global_load_async_to_lds_b128 %0, %1, %2"
               :: "v"(lds_addr), "v"(voff), "s"(sbase) : "memory");
}
__device__ __forceinline__ void wait_async0() {
  asm volatile("s_wait_asynccnt 0" ::: "memory");
}

__device__ __forceinline__ v16bf frag16(const unsigned short* base, int i0, int i1) {
  us8 lo = *(const us8*)(base + i0);
  us8 hi = *(const us8*)(base + i1);
  us16 v = __builtin_shufflevector(lo, hi, 0,1,2,3,4,5,6,7,8,9,10,11,12,13,14,15);
  return __builtin_bit_cast(v16bf, v);
}

// ---------------------------------------------------------------- mask scan
__global__ __launch_bounds__(1024)
void mask_scan(const int* __restrict__ mask, int* __restrict__ perm,
               int* __restrict__ counts) {
  __shared__ int sh[1024];
  __shared__ int totsh;
  const int t = threadIdx.x;
  int m[8]; int cnt = 0;
#pragma unroll
  for (int j = 0; j < 8; ++j) { m[j] = (mask[t * 8 + j] != 0); cnt += m[j]; }
  sh[t] = cnt;
  __syncthreads();
  for (int off = 1; off < 1024; off <<= 1) {
    int v = (t >= off) ? sh[t - off] : 0;
    __syncthreads();
    sh[t] += v;
    __syncthreads();
  }
  int incl = sh[t];
  if (t == 1023) totsh = incl;
  __syncthreads();
  const int nW = totsh;
  int one  = incl - cnt;       // masked tokens before mine
  int zero = t * 8 - one;      // unmasked tokens before mine
#pragma unroll
  for (int j = 0; j < 8; ++j) {
    int idx = t * 8 + j;
    if (m[j]) perm[one++] = idx;
    else      perm[nW + zero++] = idx;
  }
  if (t == 0) counts[0] = nW;
}

// --------------------------------------------------------------- x -> bf16
__global__ __launch_bounds__(256)
void cvt_bf16(const float* __restrict__ src, unsigned short* __restrict__ dst) {
  int t = blockIdx.x * 256 + threadIdx.x;
  dst[t] = f2bf(src[t]);
}

// ---- pack fp32 weight [K][N] into bf16 WMMA-B tiles of 16(N) x 32(K) -----
// tile = n_tile*KT + k_tile; within tile: n_in (16 rows) x 16 dwords, dword
// k2 = {bf16 W[2k2][n] , bf16 W[2k2+1][n]}  (lo = even K, matching the
// wave32 16-bit B-fragment: a VGPR holds a K pair; lanes 0-15 K=0..7,16..23).
__global__ __launch_bounds__(256)
void pack_w(const float* __restrict__ W, unsigned short* __restrict__ P,
            int K, int N) {
  int u    = blockIdx.x * 256 + threadIdx.x;
  int KT   = K >> 5;
  int nin  = u & 15;
  int k2   = (u >> 4) & 15;
  int tile = u >> 8;                 // = n_tile*KT + k_tile
  int kt   = tile % KT;
  int nt   = tile / KT;
  int k = kt * 32 + k2 * 2;
  int n = nt * 16 + nin;
  unsigned short lo = f2bf(W[(size_t)k * N + n]);
  unsigned short hi = f2bf(W[(size_t)(k + 1) * N + n]);
  ((unsigned*)P)[(size_t)(tile * 16 + nin) * 16 + k2] =
      ((unsigned)hi << 16) | (unsigned)lo;
}

// ---------------- stage 1: h[r,:] = silu(x_g @ B1) * (x_g @ B3) -----------
// block tile 128(M) x 64(N), 8 waves as 4x2, each wave 32x32 (2x2 WMMA tiles)
__global__ __launch_bounds__(256)
void ffn_gateup(const unsigned short* __restrict__ xb,
                const unsigned short* __restrict__ w1p,
                const unsigned short* __restrict__ w3p,
                const unsigned short* __restrict__ u1p,
                const unsigned short* __restrict__ u3p,
                const int* __restrict__ perm,
                const int* __restrict__ counts,
                unsigned short* __restrict__ hbuf) {
  const int nW     = counts[0];
  const int branch = blockIdx.z;
  const int brBase = branch ? nW : 0;
  const int brRows = branch ? (kTok - nW) : nW;
  const int row0   = blockIdx.y * 128;
  if (row0 >= brRows) return;
  const int n0 = blockIdx.x * 64;
  const unsigned short* B1 = branch ? u1p : w1p;
  const unsigned short* B3 = branch ? u3p : w3p;

  __shared__ __align__(16) unsigned short Asm[2][128 * 32];
  __shared__ __align__(16) unsigned short B1sm[2][64 * 32];
  __shared__ __align__(16) unsigned short B3sm[2][64 * 32];
  __shared__ int rowTok[128];
  __shared__ unsigned char rowOK[128];

  const int tid = threadIdx.x, lane = tid & 31, wave = tid >> 5;
  const int wm = wave >> 1, wn = wave & 1;

  if (tid < 128) {
    int local = row0 + tid;
    int ok = local < brRows;
    int rc = brBase + (ok ? local : (brRows - 1));
    rowTok[tid] = perm[rc];
    rowOK[tid] = (unsigned char)ok;
  }
  __syncthreads();

  const int KT = kH / 32;
  const unsigned long long xbB = (unsigned long long)(size_t)xb;
  const unsigned long long b1B = (unsigned long long)(size_t)B1;
  const unsigned long long b3B = (unsigned long long)(size_t)B3;

  auto issue = [&](int kt, int buf) {
    const int k0 = kt * 32;
#pragma unroll
    for (int t = 0; t < 2; ++t) {                 // A: 8KB -> 2 b128/thread
      int c = tid + t * 256;
      int row = c >> 2, seg = c & 3;
      unsigned lds = lds_off(&Asm[buf][0]) + (unsigned)(row * 64 + seg * 16);
      unsigned go  = (unsigned)((rowTok[row] * kH + k0 + seg * 8) * 2);
      async_ld_b128(lds, go, xbB);
    }
    {                                             // B1/B3: 4KB -> 1 each
      int sub = tid >> 6, w = tid & 63;
      unsigned rel = (unsigned)(sub * 1024 + w * 16);
      unsigned go  = (unsigned)(((((n0 >> 4) + sub) * KT) + kt) * 1024 + w * 16);
      async_ld_b128(lds_off(&B1sm[buf][0]) + rel, go, b1B);
      async_ld_b128(lds_off(&B3sm[buf][0]) + rel, go, b3B);
    }
  };

  v8f accG[2][2] = {}; v8f accU[2][2] = {};
  issue(0, 0);
  for (int kt = 0; kt < KT; ++kt) {
    const int buf = kt & 1;
    wait_async0();
    __syncthreads();
    if (kt + 1 < KT) issue(kt + 1, buf ^ 1);

    v16bf afr[2];
#pragma unroll
    for (int mi = 0; mi < 2; ++mi) {
      int row = wm * 32 + mi * 16 + (lane & 15);
      int hb = row * 32 + (lane >> 4) * 8;        // halves 0..7 / 16..23 pattern
      afr[mi] = frag16(&Asm[buf][0], hb, hb + 16);
    }
#pragma unroll
    for (int nf = 0; nf < 2; ++nf) {
      int sub = wn * 2 + nf;
      int hb = sub * 512 + (lane & 15) * 32 + (lane >> 4) * 8;
      v16bf bg = frag16(&B1sm[buf][0], hb, hb + 16);
      v16bf bu = frag16(&B3sm[buf][0], hb, hb + 16);
#pragma unroll
      for (int mi = 0; mi < 2; ++mi) {
        accG[mi][nf] = __builtin_amdgcn_wmma_f32_16x16x32_bf16(
            false, afr[mi], false, bg, (short)0, accG[mi][nf], false, false);
        accU[mi][nf] = __builtin_amdgcn_wmma_f32_16x16x32_bf16(
            false, afr[mi], false, bu, (short)0, accU[mi][nf], false, false);
      }
    }
  }

  // epilogue: silu(gate)*up -> bf16 h in permuted row order
#pragma unroll
  for (int mi = 0; mi < 2; ++mi)
#pragma unroll
    for (int nf = 0; nf < 2; ++nf)
#pragma unroll
      for (int g = 0; g < 8; ++g) {
        int lr = wm * 32 + mi * 16 + g + ((lane >> 4) << 3);
        if (!rowOK[lr]) continue;
        float a1 = accG[mi][nf][g], a3 = accU[mi][nf][g];
        float s = (a1 / (1.0f + __expf(-a1))) * a3;
        size_t r = (size_t)(brBase + row0 + lr);
        int n = n0 + wn * 32 + nf * 16 + (lane & 15);
        hbuf[r * kI + n] = f2bf(s);
      }
}

// ---------------- stage 2: out[perm[r],:] = h[r,:] @ B2 -------------------
__global__ __launch_bounds__(256)
void ffn_down(const unsigned short* __restrict__ hbuf,
              const unsigned short* __restrict__ w2p,
              const unsigned short* __restrict__ u2p,
              const int* __restrict__ perm,
              const int* __restrict__ counts,
              float* __restrict__ out) {
  const int nW     = counts[0];
  const int branch = blockIdx.z;
  const int brBase = branch ? nW : 0;
  const int brRows = branch ? (kTok - nW) : nW;
  const int row0   = blockIdx.y * 128;
  if (row0 >= brRows) return;
  const int n0 = blockIdx.x * 64;
  const unsigned short* Bp = branch ? u2p : w2p;

  __shared__ __align__(16) unsigned short Asm[2][128 * 32];
  __shared__ __align__(16) unsigned short Bsm[2][64 * 32];
  __shared__ int rowTok[128];
  __shared__ int rowR[128];
  __shared__ unsigned char rowOK[128];

  const int tid = threadIdx.x, lane = tid & 31, wave = tid >> 5;
  const int wm = wave >> 1, wn = wave & 1;

  if (tid < 128) {
    int local = row0 + tid;
    int ok = local < brRows;
    int rc = brBase + (ok ? local : (brRows - 1));
    rowTok[tid] = perm[rc];
    rowR[tid] = rc;
    rowOK[tid] = (unsigned char)ok;
  }
  __syncthreads();

  const int KT = kI / 32;
  const unsigned long long hB = (unsigned long long)(size_t)hbuf;
  const unsigned long long bB = (unsigned long long)(size_t)Bp;

  auto issue = [&](int kt, int buf) {
    const int k0 = kt * 32;
#pragma unroll
    for (int t = 0; t < 2; ++t) {
      int c = tid + t * 256;
      int row = c >> 2, seg = c & 3;
      unsigned lds = lds_off(&Asm[buf][0]) + (unsigned)(row * 64 + seg * 16);
      unsigned go  = (unsigned)((rowR[row] * kI + k0 + seg * 8) * 2);
      async_ld_b128(lds, go, hB);
    }
    {
      int sub = tid >> 6, w = tid & 63;
      unsigned rel = (unsigned)(sub * 1024 + w * 16);
      unsigned go  = (unsigned)(((((n0 >> 4) + sub) * KT) + kt) * 1024 + w * 16);
      async_ld_b128(lds_off(&Bsm[buf][0]) + rel, go, bB);
    }
  };

  v8f acc[2][2] = {};
  issue(0, 0);
  for (int kt = 0; kt < KT; ++kt) {
    const int buf = kt & 1;
    wait_async0();
    __syncthreads();
    if (kt + 1 < KT) issue(kt + 1, buf ^ 1);

    v16bf afr[2];
#pragma unroll
    for (int mi = 0; mi < 2; ++mi) {
      int row = wm * 32 + mi * 16 + (lane & 15);
      int hb = row * 32 + (lane >> 4) * 8;
      afr[mi] = frag16(&Asm[buf][0], hb, hb + 16);
    }
#pragma unroll
    for (int nf = 0; nf < 2; ++nf) {
      int sub = wn * 2 + nf;
      int hb = sub * 512 + (lane & 15) * 32 + (lane >> 4) * 8;
      v16bf bf = frag16(&Bsm[buf][0], hb, hb + 16);
#pragma unroll
      for (int mi = 0; mi < 2; ++mi)
        acc[mi][nf] = __builtin_amdgcn_wmma_f32_16x16x32_bf16(
            false, afr[mi], false, bf, (short)0, acc[mi][nf], false, false);
    }
  }

#pragma unroll
  for (int mi = 0; mi < 2; ++mi)
#pragma unroll
    for (int nf = 0; nf < 2; ++nf)
#pragma unroll
      for (int g = 0; g < 8; ++g) {
        int lr = wm * 32 + mi * 16 + g + ((lane >> 4) << 3);
        if (!rowOK[lr]) continue;
        int n = n0 + wn * 32 + nf * 16 + (lane & 15);
        out[(size_t)rowTok[lr] * kH + n] = acc[mi][nf][g];
      }
}

// -------------------------------------------------------------------------
extern "C" void kernel_launch(void* const* d_in, const int* in_sizes, int n_in,
                              void* d_out, int out_size, void* d_ws, size_t ws_size,
                              hipStream_t stream) {
  const float* x  = (const float*)d_in[0];
  const float* w1 = (const float*)d_in[1];
  const float* w2 = (const float*)d_in[2];
  const float* w3 = (const float*)d_in[3];
  const float* u1 = (const float*)d_in[4];
  const float* u2 = (const float*)d_in[5];
  const float* u3 = (const float*)d_in[6];
  const int* mask = (const int*)d_in[7];
  float* out = (float*)d_out;

  char* ws = (char*)d_ws;
  size_t o = 0;
  auto alloc = [&](size_t bytes) {
    void* p = ws + o;
    o = (o + bytes + 255) & ~(size_t)255;
    return p;
  };
  int* perm   = (int*)alloc((size_t)kTok * 4);
  int* counts = (int*)alloc(256);
  unsigned short* xb  = (unsigned short*)alloc((size_t)kTok * kH * 2);
  unsigned short* w1p = (unsigned short*)alloc((size_t)kH * kI * 2);
  unsigned short* w3p = (unsigned short*)alloc((size_t)kH * kI * 2);
  unsigned short* u1p = (unsigned short*)alloc((size_t)kH * kI * 2);
  unsigned short* u3p = (unsigned short*)alloc((size_t)kH * kI * 2);
  unsigned short* w2p = (unsigned short*)alloc((size_t)kI * kH * 2);
  unsigned short* u2p = (unsigned short*)alloc((size_t)kI * kH * 2);
  unsigned short* hb  = (unsigned short*)alloc((size_t)kTok * kI * 2);
  (void)ws_size; (void)in_sizes; (void)n_in; (void)out_size;

  mask_scan<<<1, 1024, 0, stream>>>(mask, perm, counts);
  cvt_bf16<<<(kTok * kH) / 256, 256, 0, stream>>>(x, xb);

  const int packBlocks = (kH * kI) / 512;   // dwords / 256
  pack_w<<<packBlocks, 256, 0, stream>>>(w1, w1p, kH, kI);
  pack_w<<<packBlocks, 256, 0, stream>>>(w3, w3p, kH, kI);
  pack_w<<<packBlocks, 256, 0, stream>>>(u1, u1p, kH, kI);
  pack_w<<<packBlocks, 256, 0, stream>>>(u3, u3p, kH, kI);
  pack_w<<<packBlocks, 256, 0, stream>>>(w2, w2p, kI, kH);
  pack_w<<<packBlocks, 256, 0, stream>>>(u2, u2p, kI, kH);

  ffn_gateup<<<dim3(kI / 64, kTok / 128, 2), 256, 0, stream>>>(
      xb, w1p, w3p, u1p, u3p, perm, counts, hb);
  ffn_down<<<dim3(kH / 64, kTok / 128, 2), 256, 0, stream>>>(
      hb, w2p, u2p, perm, counts, out);
}
